// SimpleQDGAT_12043088298061
// MI455X (gfx1250) — matrix operand order repaired and checked by reading, not compiled, not run on hardware
//
#include <hip/hip_runtime.h>

#define B_ 8
#define N_ 512
#define H_ 256
#define T_ 4

typedef __attribute__((ext_vector_type(16))) __bf16 v16bf;
typedef __attribute__((ext_vector_type(8)))  __bf16 bf16x8;
typedef __attribute__((ext_vector_type(8)))  float  v8f;

__device__ __forceinline__ unsigned short f2bfu(float f) {
  unsigned int u = __builtin_bit_cast(unsigned int, f);
  u += 0x7FFFu + ((u >> 16) & 1u);                 // round-to-nearest-even
  return (unsigned short)(u >> 16);
}

__device__ __forceinline__ float block_sum(float v, float* red) {
  int tid = threadIdx.x;
  red[tid] = v; __syncthreads();
  for (int s = 128; s > 0; s >>= 1) { if (tid < s) red[tid] += red[tid + s]; __syncthreads(); }
  float r = red[0]; __syncthreads();
  return r;
}
__device__ __forceinline__ float block_max(float v, float* red) {
  int tid = threadIdx.x;
  red[tid] = v; __syncthreads();
  for (int s = 128; s > 0; s >>= 1) { if (tid < s) red[tid] = fmaxf(red[tid], red[tid + s]); __syncthreads(); }
  float r = red[0]; __syncthreads();
  return r;
}

// C[z] = A[z] @ W[z];  A given as split-K pair (k < K0 -> A0, else A1) so that
// concat(enc, node) / concat(enc, X_t) never has to be materialized.
// block = 256 threads = 8 waves; tile = 128 rows x 16 cols; KSTEP = 64 (2 bf16 WMMAs/stage).
// LDS layout is 16B-aligned so every fragment is two ds_load_b128; all hot-loop
// addressing is strength-reduced to pointer increments (no per-load 64-bit muls).
__global__ __launch_bounds__(256) void wmma_gemm_kernel(
    const float* __restrict__ A0, const float* __restrict__ A1,
    int K0, int K, int lda0, int lda1,
    const float* __restrict__ W, int ldw,
    float* __restrict__ C, int Nn,
    long sA0, long sA1, long sW, long sC)
{
  __shared__ __attribute__((aligned(16))) __bf16 As[128][72];  // 64 + 8 pad (stride 144B)
  __shared__ __attribute__((aligned(16))) __bf16 Wt[16][72];   // transposed: [n][k]

  const int z = blockIdx.z;
  A0 += (long)z * sA0;  A1 += (long)z * sA1;
  W  += (long)z * sW;   C  += (long)z * sC;

  const int m0   = blockIdx.x * 128;
  const int n0   = blockIdx.y * 16;
  const int tid  = threadIdx.x;
  const int lane = tid & 31;
  const int wave = tid >> 5;
  const int hh   = lane >> 4;      // half-wave (0/1)
  const int ml   = lane & 15;

  const int kkA = (2 * tid) & 63;        // per-thread A column pair (loop-invariant)
  const int rA  = tid >> 5;              // A row base
  const int nB  = tid & 15;              // per-thread W column
  const int kB  = 2 * (tid >> 4);        // per-thread W k pair

  v8f acc = {};

  for (int k0 = 0; k0 < K; k0 += 64) {
    // ---- stage 128x64 A tile: float2 loads (coalesced), packed b32 LDS stores.
    // One 32-bit base computation per stage; unrolled loop advances by 8*ld.
    {
      int kg = k0 + kkA;
      const float* pA; int ld;
      if (kg < K0) { pA = A0 + kg;        ld = lda0; }
      else         { pA = A1 + (kg - K0); ld = lda1; }
      pA += (size_t)((m0 + rA) * ld);
      const size_t step = (size_t)(8 * ld);
#pragma unroll
      for (int i = 0; i < 16; ++i) {
        float2 v = *(const float2*)pA;
        pA += step;
        unsigned int pk = (unsigned int)f2bfu(v.x) | ((unsigned int)f2bfu(v.y) << 16);
        *(unsigned int*)&As[rA + 8 * i][kkA] = pk;
      }
      // prefetch next A tile (global_prefetch_b8)
      int kg2 = k0 + 64 + kkA;
      if (kg2 < K) {
        const float* pf = (kg2 < K0) ? A0 + (size_t)((m0 + rA) * lda0) + kg2
                                     : A1 + (size_t)((m0 + rA) * lda1) + (kg2 - K0);
        __builtin_prefetch(pf, 0, 1);
      }
    }
    // ---- stage 64x16 W tile transposed into Wt[n][k], packed b32 stores
    {
      const float* pW = W + (size_t)((k0 + kB) * ldw) + (n0 + nB);
      const size_t stepW = (size_t)(32 * ldw);
#pragma unroll
      for (int j = 0; j < 2; ++j) {
        float v0 = pW[0];
        float v1 = pW[ldw];
        pW += stepW;
        unsigned int pk = (unsigned int)f2bfu(v0) | ((unsigned int)f2bfu(v1) << 16);
        *(unsigned int*)&Wt[nB][kB + 32 * j] = pk;
      }
    }
    __syncthreads();

    // ---- 2 WMMAs; every fragment = two ds_load_b128
    const int ar = wave * 16 + ml;
#pragma unroll
    for (int kh = 0; kh < 2; ++kh) {
      int ka = kh * 32 + hh * 8;
      bf16x8 a_lo = *(const bf16x8*)&As[ar][ka];
      bf16x8 a_hi = *(const bf16x8*)&As[ar][ka + 16];
      int kb = kh * 32 + hh * 8;
      bf16x8 b_lo = *(const bf16x8*)&Wt[ml][kb];
      bf16x8 b_hi = *(const bf16x8*)&Wt[ml][kb + 16];
      v16bf af = __builtin_shufflevector(a_lo, a_hi, 0,1,2,3,4,5,6,7,8,9,10,11,12,13,14,15);
      v16bf bf = __builtin_shufflevector(b_lo, b_hi, 0,1,2,3,4,5,6,7,8,9,10,11,12,13,14,15);
      acc = __builtin_amdgcn_wmma_f32_16x16x32_bf16(false, af, false, bf,
                                                    (short)0, acc, false, false);
    }
    __syncthreads();
  }

  // C/D layout: VGPR v -> row (v + 8*half), col = lane&15
#pragma unroll
  for (int v = 0; v < 8; ++v) {
    int r = m0 + wave * 16 + v + 8 * hh;
    C[(long)r * Nn + n0 + ml] = acc[v];
  }
}

// h_q = elu(q @ W_fc)   [B, 2H]
__global__ void hq_kernel(const float* __restrict__ q, const float* __restrict__ Wfc,
                          float* __restrict__ hq)
{
  int idx = blockIdx.x * blockDim.x + threadIdx.x;   // B*2H
  int b = idx >> 9, j = idx & 511;
  float s = 0.f;
  for (int i = 0; i < H_; ++i) s += q[b * H_ + i] * Wfc[i * (2 * H_) + j];
  hq[idx] = s > 0.f ? s : (__expf(s) - 1.0f);
}

// m = h_q @ W_dc[t]   [B, H]
__global__ void mt_kernel(const float* __restrict__ hq, const float* __restrict__ Wdc_t,
                          float* __restrict__ m)
{
  int idx = blockIdx.x * blockDim.x + threadIdx.x;   // B*H
  int b = idx >> 8, h = idx & 255;
  float s = 0.f;
  for (int j = 0; j < 2 * H_; ++j) s += hq[b * 2 * H_ + j] * Wdc_t[j * H_ + h];
  m[idx] = s;
}

// mq/mk/mv = m @ {W_qc, W_kc, W_vc}
__global__ void mproj_kernel(const float* __restrict__ m,
                             const float* __restrict__ Wq, const float* __restrict__ Wk,
                             const float* __restrict__ Wv,
                             float* __restrict__ mq, float* __restrict__ mk,
                             float* __restrict__ mv)
{
  int idx = blockIdx.x * blockDim.x + threadIdx.x;   // 3*B*H
  int w = idx / (B_ * H_), rem = idx % (B_ * H_);
  int b = rem >> 8, h = rem & 255;
  const float* W = (w == 0) ? Wq : (w == 1) ? Wk : Wv;
  float s = 0.f;
  for (int i = 0; i < H_; ++i) s += m[b * H_ + i] * W[i * H_ + h];
  ((w == 0) ? mq : (w == 1) ? mk : mv)[rem] = s;
}

// X = LayerNorm(Y * mvec_broadcast) ; mvec == nullptr -> plain LayerNorm
__global__ __launch_bounds__(256) void scale_ln_kernel(
    const float* __restrict__ Y, const float* __restrict__ mvec,
    const float* __restrict__ g, const float* __restrict__ bb,
    float eps, float* __restrict__ X)
{
  __shared__ float red[256];
  int row = blockIdx.x;
  int b = row >> 9;                 // row / N_
  int h = threadIdx.x;
  float v = Y[(long)row * H_ + h];
  if (mvec) v *= mvec[b * H_ + h];
  float mean = block_sum(v, red) * (1.0f / H_);
  float d = v - mean;
  float var = block_sum(d * d, red) * (1.0f / H_);
  float inv = rsqrtf(var + eps);
  X[(long)row * H_ + h] = d * inv * g[h] + bb[h];
}

// a[row,k] = x_q[row]·att_wx[k] ; c[row,k] = x_k[row]·att_wy[k]
__global__ void ac_kernel(const float* __restrict__ xq, const float* __restrict__ xk,
                          const float* __restrict__ wx, const float* __restrict__ wy,
                          float* __restrict__ a, float* __restrict__ c)
{
  int idx = blockIdx.x * blockDim.x + threadIdx.x;   // 2*B*N*9
  const int total = B_ * N_ * 9;
  int which = idx / total, rem = idx % total;
  int row = rem / 9, k = rem % 9;
  const float* x = which ? xk : xq;
  const float* w = which ? wy : wx;
  float s = 0.f;
  for (int h = 0; h < H_; ++h) s += x[(long)row * H_ + h] * w[k * H_ + h];
  (which ? c : a)[rem] = s;
}

// Per (b,i) row: 9-head leaky-relu scores, both masked softmaxes, combined p row.
// masked_softmax(s, m): p = softmax(s*m)*m ; p/(Σp+1e-13) == e*m/(Q + 1e-13*Z)
__global__ __launch_bounds__(256) void score_kernel(
    const float* __restrict__ a, const float* __restrict__ c,
    const float* __restrict__ attb,
    const int* __restrict__ mask_e, const int* __restrict__ mask_nn,
    float* __restrict__ p)
{
  __shared__ float t_nn[512], w_nn[512], t_en[512], w_en[512];
  __shared__ float red[256];
  __shared__ float sa[9], sb9[9];

  int row = blockIdx.x;              // b*N + i
  int tid = threadIdx.x;
  if (tid < 9) { sa[tid] = a[(long)row * 9 + tid]; sb9[tid] = attb[tid]; }
  __syncthreads();

  int b = row >> 9;
  long base = (long)row * N_;        // (b,i,*) in [B,N,N]
  const long BNN = (long)B_ * N_ * N_;

  for (int jj = 0; jj < 2; ++jj) {
    int j = tid + jj * 256;
    long cb = ((long)b * N_ + j) * 9;
    float nn = 0.f, msum = 0.f;
#pragma unroll
    for (int k = 0; k < 8; ++k) {
      float sc = sa[k] + c[cb + k] + sb9[k];
      sc = sc > 0.f ? sc : 0.01f * sc;
      if (mask_nn[k * BNN + base + j]) { nn += sc; msum += 1.f; }
    }
    float s8 = sa[8] + c[cb + 8] + sb9[8];
    s8 = s8 > 0.f ? s8 : 0.01f * s8;
    float me = (mask_e[base + j] != 0) ? 1.f : 0.f;
    t_nn[j] = nn * msum;  w_nn[j] = msum;   // softmax input = score*mask (mask = Σmasks, 0..8)
    t_en[j] = s8 * me;    w_en[j] = me;
  }
  __syncthreads();

  float mx_nn = block_max(fmaxf(t_nn[tid], t_nn[tid + 256]), red);
  float mx_en = block_max(fmaxf(t_en[tid], t_en[tid + 256]), red);

  float zn = 0.f, qn = 0.f, ze = 0.f, qe = 0.f;
  for (int jj = 0; jj < 2; ++jj) {
    int j = tid + jj * 256;
    float e1 = __expf(t_nn[j] - mx_nn); t_nn[j] = e1; zn += e1; qn += e1 * w_nn[j];
    float e2 = __expf(t_en[j] - mx_en); t_en[j] = e2; ze += e2; qe += e2 * w_en[j];
  }
  zn = block_sum(zn, red);
  qn = block_sum(qn, red);
  ze = block_sum(ze, red);
  qe = block_sum(qe, red);

  float inn = 1.0f / (qn + 1e-13f * zn);
  float ien = 1.0f / (qe + 1e-13f * ze);
  for (int jj = 0; jj < 2; ++jj) {
    int j = tid + jj * 256;
    p[base + j] = t_nn[j] * w_nn[j] * inn + t_en[j] * w_en[j] * ien;
  }
}

extern "C" void kernel_launch(void* const* d_in, const int* in_sizes, int n_in,
                              void* d_out, int out_size, void* d_ws, size_t ws_size,
                              hipStream_t stream)
{
  (void)in_sizes; (void)n_in; (void)out_size; (void)ws_size;
  const float* node   = (const float*)d_in[0];
  const float* qemb   = (const float*)d_in[1];
  const int*   mask_e = (const int*)d_in[2];
  const int*   mask_nn= (const int*)d_in[3];
  const float* W_fc   = (const float*)d_in[4];
  const float* W_dc   = (const float*)d_in[5];
  const float* W_qv   = (const float*)d_in[6];
  const float* W_kv   = (const float*)d_in[7];
  const float* W_vv   = (const float*)d_in[8];
  const float* W_qc   = (const float*)d_in[9];
  const float* W_kc   = (const float*)d_in[10];
  const float* W_vc   = (const float*)d_in[11];
  const float* W_u    = (const float*)d_in[12];
  const float* att_wx = (const float*)d_in[13];
  const float* att_wy = (const float*)d_in[14];
  const float* att_b  = (const float*)d_in[15];
  const float* ln_g   = (const float*)d_in[16];
  const float* ln_b   = (const float*)d_in[17];
  float* enc = (float*)d_out;

  float* ws = (float*)d_ws;
  size_t cur = 0;
  auto alloc = [&](size_t n) { float* r = ws + cur; cur += n; return r; };
  const int BN = B_ * N_;
  float* hq  = alloc((size_t)B_ * 2 * H_);
  float* mb  = alloc((size_t)B_ * H_);
  float* mqv = alloc((size_t)B_ * H_);
  float* mkv = alloc((size_t)B_ * H_);
  float* mvv = alloc((size_t)B_ * H_);
  float* Yq  = alloc((size_t)BN * H_);
  float* Yk  = alloc((size_t)BN * H_);
  float* Yv  = alloc((size_t)BN * H_);
  float* xq  = alloc((size_t)BN * H_);
  float* xk  = alloc((size_t)BN * H_);
  float* xv  = alloc((size_t)BN * H_);
  float* av  = alloc((size_t)BN * 9);
  float* cv  = alloc((size_t)BN * 9);
  float* pw  = alloc((size_t)B_ * N_ * N_);
  float* Xt  = alloc((size_t)BN * H_);
  float* U   = alloc((size_t)BN * H_);

  // enc <- node_embedding
  hipMemcpyAsync(enc, node, (size_t)BN * H_ * sizeof(float),
                 hipMemcpyDeviceToDevice, stream);

  hq_kernel<<<(B_ * 2 * H_) / 256, 256, 0, stream>>>(qemb, W_fc, hq);

  for (int t = 0; t < T_; ++t) {
    mt_kernel<<<(B_ * H_) / 256, 256, 0, stream>>>(hq, W_dc + (size_t)t * 2 * H_ * H_, mb);
    mproj_kernel<<<(3 * B_ * H_) / 256, 256, 0, stream>>>(mb, W_qc, W_kc, W_vc, mqv, mkv, mvv);

    dim3 g1(BN / 128, H_ / 16, 1);
    wmma_gemm_kernel<<<g1, 256, 0, stream>>>(enc, node, H_, 2 * H_, H_, H_, W_qv, H_, Yq, H_, 0, 0, 0, 0);
    wmma_gemm_kernel<<<g1, 256, 0, stream>>>(enc, node, H_, 2 * H_, H_, H_, W_kv, H_, Yk, H_, 0, 0, 0, 0);
    wmma_gemm_kernel<<<g1, 256, 0, stream>>>(enc, node, H_, 2 * H_, H_, H_, W_vv, H_, Yv, H_, 0, 0, 0, 0);

    scale_ln_kernel<<<BN, 256, 0, stream>>>(Yq, mqv, ln_g + 0 * H_, ln_b + 0 * H_, 1e-12f, xq);
    scale_ln_kernel<<<BN, 256, 0, stream>>>(Yk, mkv, ln_g + 1 * H_, ln_b + 1 * H_, 1e-12f, xk);
    scale_ln_kernel<<<BN, 256, 0, stream>>>(Yv, mvv, ln_g + 2 * H_, ln_b + 2 * H_, 1e-12f, xv);

    ac_kernel<<<(2 * BN * 9) / 256, 256, 0, stream>>>(xq, xk, att_wx, att_wy, av, cv);

    score_kernel<<<BN, 256, 0, stream>>>(av, cv, att_b, mask_e, mask_nn, pw);

    // X_t = (p_nn + p_en) @ x_v   (batched over B)
    dim3 g2(N_ / 128, H_ / 16, B_);
    wmma_gemm_kernel<<<g2, 256, 0, stream>>>(pw, pw, N_, N_, N_, N_, xv, H_, Xt, H_,
                                             (long)N_ * N_, (long)N_ * N_,
                                             (long)N_ * H_, (long)N_ * H_);

    // enc = LN(concat(enc, X_t) @ W_u)
    wmma_gemm_kernel<<<g1, 256, 0, stream>>>(enc, Xt, H_, 2 * H_, H_, H_, W_u, H_, U, H_, 0, 0, 0, 0);
    scale_ln_kernel<<<BN, 256, 0, stream>>>(U, nullptr, ln_g + 3 * H_, ln_b + 3 * H_, 1e-5f, enc);
  }
}
